// Channel_attention_22677427322968
// MI455X (gfx1250) — matrix operand order repaired and checked by reading, not compiled, not run on hardware
//
#include <hip/hip_runtime.h>

typedef __attribute__((ext_vector_type(16))) _Float16 v16h;
typedef __attribute__((ext_vector_type(8)))  _Float16 v8h;
typedef __attribute__((ext_vector_type(8)))  float    v8f;

// LDS row strides (halves): 72 halves = 144 B -> 36-bank lane stride, all 16
// lanes of a b128 fragment load hit distinct banks.
#define SQS 72   // Q tile    [64 rows][64 C]
#define SKS 72   // K chunk   [64 keys][64 C]
#define SKTS 72  // K chunk^T [64 C ][64 keys]
#define SPS 72   // P scratch [16 rows][64 keys] per wave

__device__ __forceinline__ v16h ld_frag(const _Float16* p0, const _Float16* p1) {
  v8h lo = *(const v8h*)p0;
  v8h hh = *(const v8h*)p1;
  v16h r;
#pragma unroll
  for (int i = 0; i < 8; ++i) { r[i] = lo[i]; r[i + 8] = hh[i]; }
  return r;
}

// 16-lane butterfly reductions on the VALU pipe via DPP16 (no DS traffic).
// xor1 = quad_perm(1,0,3,2)=0xB1, xor2 = quad_perm(2,3,0,1)=0x4E,
// xor4 = row_half_mirror=0x141, xor8 = row_mirror=0x140.
template <int CTRL>
__device__ __forceinline__ float dpp_f(float v) {
  return __int_as_float(
      __builtin_amdgcn_update_dpp(0, __float_as_int(v), CTRL, 0xF, 0xF, true));
}
__device__ __forceinline__ float red16max(float v) {
  v = fmaxf(v, dpp_f<0xB1>(v));
  v = fmaxf(v, dpp_f<0x4E>(v));
  v = fmaxf(v, dpp_f<0x141>(v));
  v = fmaxf(v, dpp_f<0x140>(v));
  return v;
}
__device__ __forceinline__ float red16sum(float v) {
  v += dpp_f<0xB1>(v);
  v += dpp_f<0x4E>(v);
  v += dpp_f<0x141>(v);
  v += dpp_f<0x140>(v);
  return v;
}

#define WMMA_F16(A, B, Cacc) \
  __builtin_amdgcn_wmma_f32_16x16x32_f16(false, (A), false, (B), (short)0, (Cacc), false, false)

__global__ __launch_bounds__(128) void chan_attn_kernel(const float* __restrict__ x,
                                                        const float* __restrict__ gamma,
                                                        float* __restrict__ out) {
  constexpr int N = 4096, C = 64;

  __shared__ __align__(16) _Float16 sQ [64 * SQS];
  __shared__ __align__(16) _Float16 sK [64 * SKS];
  __shared__ __align__(16) _Float16 sKT[64 * SKTS];
  __shared__ __align__(16) _Float16 sP [4 * 16 * SPS];

  const int tid    = threadIdx.x;
  const int wave   = tid >> 5;
  const int lane   = tid & 31;
  const int lane16 = lane & 15;
  const int hi     = lane >> 4;      // 0 for lanes 0-15, 1 for lanes 16-31

  const int b     = blockIdx.x >> 6;           // 2 batches
  const int qrow0 = (blockIdx.x & 63) * 64;    // 64 rows per WG
  const float* xb = x + (size_t)b * N * C;
  const float* xq = xb + (size_t)qrow0 * C;

  // ---- stage Q tile (64x64 f32 -> f16 in LDS) ----
#pragma unroll
  for (int i = 0; i < 8; ++i) {
    int idx = i * 128 + tid;         // 0..1023 float4's
    int row = idx >> 4;
    int c4  = (idx & 15) << 2;
    const float4 v = *(const float4*)(xq + row * C + c4);
    _Float16* dq = &sQ[row * SQS + c4];
    dq[0] = (_Float16)v.x; dq[1] = (_Float16)v.y;
    dq[2] = (_Float16)v.z; dq[3] = (_Float16)v.w;
  }
  __syncthreads();

  // ---- persistent A fragments: this wave's 16 Q rows, K(=C) split 0..31/32..63 ----
  const int arow = wave * 16 + lane16;
  const _Float16* qp = &sQ[arow * SQS + hi * 16];
  const v16h a_lo = ld_frag(qp,      qp + 8);    // C 0..31
  const v16h a_hi = ld_frag(qp + 32, qp + 40);   // C 32..63

  v8f o0 = {}, o1 = {}, o2 = {}, o3 = {};
  float mrow[8], lrow[8];
#pragma unroll
  for (int r = 0; r < 8; ++r) { mrow[r] = -INFINITY; lrow[r] = 0.0f; }

  _Float16* sPw = &sP[wave * 16 * SPS];

  for (int kb = 0; kb < N; kb += 64) {
    __syncthreads();
    // ---- stage 64-key K/V chunk in both layouts ----
    const float* xk = xb + (size_t)kb * C;
#pragma unroll
    for (int i = 0; i < 8; ++i) {
      int idx = i * 128 + tid;       // 0..1023 float4's
      int kr  = idx >> 4;
      int c4  = (idx & 15) << 2;
      const float4 v = *(const float4*)(xk + kr * C + c4);
      _Float16 h0 = (_Float16)v.x, h1 = (_Float16)v.y;
      _Float16 h2 = (_Float16)v.z, h3 = (_Float16)v.w;
      _Float16* dk = &sK[kr * SKS + c4];
      dk[0] = h0; dk[1] = h1; dk[2] = h2; dk[3] = h3;
      sKT[(c4 + 0) * SKTS + kr] = h0;
      sKT[(c4 + 1) * SKTS + kr] = h1;
      sKT[(c4 + 2) * SKTS + kr] = h2;
      sKT[(c4 + 3) * SKTS + kr] = h3;
    }
    if (kb + 64 < N) {
      __builtin_prefetch(xk + 64 * C + (tid << 3), 0, 3);  // next chunk -> L2/LDS path
    }
    __syncthreads();

    // ---- GEMM1: S = Q_tile x K_chunk^T, four 16x16 tiles, f32 accum ----
    v8f z = {};
    const _Float16* k0 = &sK[( 0 + lane16) * SKS + hi * 16];
    const _Float16* k1 = &sK[(16 + lane16) * SKS + hi * 16];
    const _Float16* k2 = &sK[(32 + lane16) * SKS + hi * 16];
    const _Float16* k3 = &sK[(48 + lane16) * SKS + hi * 16];
    v8f s0 = WMMA_F16(a_lo, ld_frag(k0,      k0 + 8),  z);
    s0     = WMMA_F16(a_hi, ld_frag(k0 + 32, k0 + 40), s0);
    v8f s1 = WMMA_F16(a_lo, ld_frag(k1,      k1 + 8),  z);
    s1     = WMMA_F16(a_hi, ld_frag(k1 + 32, k1 + 40), s1);
    v8f s2 = WMMA_F16(a_lo, ld_frag(k2,      k2 + 8),  z);
    s2     = WMMA_F16(a_hi, ld_frag(k2 + 32, k2 + 40), s2);
    v8f s3 = WMMA_F16(a_lo, ld_frag(k3,      k3 + 8),  z);
    s3     = WMMA_F16(a_hi, ld_frag(k3 + 32, k3 + 40), s3);

    // ---- online softmax: row r is (r + 8*hi); DPP 16-lane reductions ----
#pragma unroll
    for (int r = 0; r < 8; ++r) {
      float smax = fmaxf(fmaxf(s0[r], s1[r]), fmaxf(s2[r], s3[r]));
      smax = red16max(smax);
      float mn = fmaxf(mrow[r], smax);
      float sc = __expf(mrow[r] - mn);
      mrow[r] = mn;
      float p0 = __expf(s0[r] - mn);
      float p1 = __expf(s1[r] - mn);
      float p2 = __expf(s2[r] - mn);
      float p3 = __expf(s3[r] - mn);
      lrow[r] = lrow[r] * sc + red16sum((p0 + p1) + (p2 + p3));
      o0[r] *= sc; o1[r] *= sc; o2[r] *= sc; o3[r] *= sc;
      _Float16* pw = &sPw[(r + 8 * hi) * SPS + lane16];
      pw[0]  = (_Float16)p0;
      pw[16] = (_Float16)p1;
      pw[32] = (_Float16)p2;
      pw[48] = (_Float16)p3;
    }

    // ---- GEMM2: O += P x V (same-wave LDS bounce transposes P into A layout) ----
    const _Float16* pp = &sPw[lane16 * SPS + hi * 16];
    v16h ap0 = ld_frag(pp,      pp + 8);    // keys kb+0..31
    v16h ap1 = ld_frag(pp + 32, pp + 40);   // keys kb+32..63
    const _Float16* v0p = &sKT[( 0 + lane16) * SKTS + hi * 16];
    const _Float16* v1p = &sKT[(16 + lane16) * SKTS + hi * 16];
    const _Float16* v2p = &sKT[(32 + lane16) * SKTS + hi * 16];
    const _Float16* v3p = &sKT[(48 + lane16) * SKTS + hi * 16];
    o0 = WMMA_F16(ap0, ld_frag(v0p,      v0p + 8),  o0);
    o0 = WMMA_F16(ap1, ld_frag(v0p + 32, v0p + 40), o0);
    o1 = WMMA_F16(ap0, ld_frag(v1p,      v1p + 8),  o1);
    o1 = WMMA_F16(ap1, ld_frag(v1p + 32, v1p + 40), o1);
    o2 = WMMA_F16(ap0, ld_frag(v2p,      v2p + 8),  o2);
    o2 = WMMA_F16(ap1, ld_frag(v2p + 32, v2p + 40), o2);
    o3 = WMMA_F16(ap0, ld_frag(v3p,      v3p + 8),  o3);
    o3 = WMMA_F16(ap1, ld_frag(v3p + 32, v3p + 40), o3);
  }

  // ---- epilogue: out = gamma * (O / l) + x ----
  const float g = gamma[0];
  float* ob = out + (size_t)b * N * C;
#pragma unroll
  for (int r = 0; r < 8; ++r) {
    const float inv = 1.0f / lrow[r];
    const size_t row = (size_t)(qrow0 + wave * 16 + r + 8 * hi);
    float acc[4] = { o0[r], o1[r], o2[r], o3[r] };
#pragma unroll
    for (int cc = 0; cc < 4; ++cc) {
      size_t idx = row * C + cc * 16 + lane16;
      ob[idx] = g * (acc[cc] * inv) + xb[idx];
    }
  }
}

extern "C" void kernel_launch(void* const* d_in, const int* in_sizes, int n_in,
                              void* d_out, int out_size, void* d_ws, size_t ws_size,
                              hipStream_t stream) {
  (void)in_sizes; (void)n_in; (void)out_size; (void)d_ws; (void)ws_size;
  const float* x     = (const float*)d_in[0];
  const float* gamma = (const float*)d_in[1];
  float* out = (float*)d_out;
  // 2 batches * (4096/64) row-blocks = 128 workgroups of 128 threads (4 waves)
  chan_attn_kernel<<<dim3(128), dim3(128), 0, stream>>>(x, gamma, out);
}
